// HGTLayer_46608985096863
// MI455X (gfx1250) — compile-verified, matrix operand order checked by reading, not run on hardware
//
#include <hip/hip_runtime.h>
#include <stdint.h>

// ---------------------------------------------------------------------------
// Types for CDNA5 WMMA (wave32): v_wmma_f32_16x16x32_bf16
// ---------------------------------------------------------------------------
typedef __attribute__((ext_vector_type(16))) __bf16 v16bf;
typedef __attribute__((ext_vector_type(8)))  float  v8f;
typedef __attribute__((ext_vector_type(4)))  unsigned int u32x4;

union Frag16 { v16bf v; u32x4 q[2]; };

__device__ __forceinline__ unsigned short f2bf(float f) {
  unsigned u = __float_as_uint(f);
  u += 0x7fffu + ((u >> 16) & 1u);           // round-to-nearest-even
  return (unsigned short)(u >> 16);
}
__device__ __forceinline__ float bf2f(unsigned short s) {
  return __uint_as_float(((unsigned)s) << 16);
}
// monotonic float<->uint mapping so atomicMax(uint) == float max (handles negatives)
__device__ __forceinline__ unsigned fkey(float f) {
  unsigned u = __float_as_uint(f);
  return (u & 0x80000000u) ? ~u : (u | 0x80000000u);
}
__device__ __forceinline__ float funkey(unsigned k) {
  unsigned u = (k & 0x80000000u) ? (k & 0x7fffffffu) : ~k;
  return __uint_as_float(u);
}

#define DIM 256
#define H 4
#define DK 64
// key(-inf) for the monotonic uint encoding: ~0x007fffff == 0xFF800000 == -inf
#define FKEY_NEG_INF 0x007fffffu

// ---------------------------------------------------------------------------
// f32 -> bf16 conversion of node features
// ---------------------------------------------------------------------------
__global__ void hgt_f2bf_kernel(const float* __restrict__ in,
                                unsigned short* __restrict__ out, int n) {
  int i = blockIdx.x * blockDim.x + threadIdx.x;
  if (i < n) out[i] = f2bf(in[i]);
}

// ---------------------------------------------------------------------------
// Fuse per-type projection weight with per-head relation transform, store
// TRANSPOSED ([col][k], bf16) for contiguous WMMA B-fragment loads.
//   WT[r][c][k] = sum_i W[s(r)][k][h*64+i] * rel[r][h][i][j],  c = h*64+j
//   bF[r][c]    = sum_i b[s(r)][h*64+i]    * rel[r][h][i][j]
// ---------------------------------------------------------------------------
__global__ void hgt_fuse_weight_kernel(const float* __restrict__ W,   // [T,256,256]
                                       const float* __restrict__ rel, // [R,H,64,64]
                                       const float* __restrict__ b,   // [T,256]
                                       unsigned short* __restrict__ WT,// [R,256,256]
                                       float* __restrict__ bF) {      // [R,256]
  int idx = blockIdx.x * blockDim.x + threadIdx.x;
  if (idx >= 3 * 256 * 256) return;
  int r = idx >> 16;
  int c = (idx >> 8) & 255;
  int k = idx & 255;
  int sType = (r == 1) ? 1 : 0;   // relations: 0:A->B, 1:B->A, 2:A->B
  int h = c >> 6, j = c & 63;
  const float* wrow = W + (size_t)sType * 65536 + (size_t)k * 256 + h * 64;
  const float* arel = rel + ((size_t)(r * 4 + h) * 64) * 64 + j;
  float acc = 0.0f;
  #pragma unroll 8
  for (int i = 0; i < 64; ++i) acc += wrow[i] * arel[(size_t)i * 64];
  WT[(size_t)r * 65536 + (size_t)c * 256 + k] = f2bf(acc);
  if (k == 0) {
    const float* brow = b + sType * 256 + h * 64;
    float bb = 0.0f;
    for (int i = 0; i < 64; ++i) bb += brow[i] * arel[(size_t)i * 64];
    bF[r * 256 + c] = bb;
  }
}

// Transpose+convert plain [T,256,256] weight to [T][col][k] bf16
__global__ void hgt_transpose_w_kernel(const float* __restrict__ W,
                                       unsigned short* __restrict__ WT) {
  int idx = blockIdx.x * blockDim.x + threadIdx.x;
  if (idx >= 2 * 65536) return;
  int t = idx >> 16, c = (idx >> 8) & 255, k = idx & 255;
  WT[idx] = f2bf(W[(size_t)t * 65536 + (size_t)k * 256 + c]);
}

// ---------------------------------------------------------------------------
// bf16 WMMA GEMM:  C[M,256] = A[M,256] @ W[256,256] + bias
//  - W passed TRANSPOSED ([col][k]); one 64-col slice staged in LDS per block
//  - block = 256 threads = 8 waves; wave w -> rows [blk*128+w*16, +16)
//  - grid = (ceil(M/128), 4 col-blocks)
//  mode 0: store bf16.  mode 1: out = sigmoid(skip)*acc + (1-sig)*hres (f32)
// ---------------------------------------------------------------------------
__global__ __launch_bounds__(256)
void hgt_gemm_bf16(const unsigned short* __restrict__ A,
                   const unsigned short* __restrict__ WT,
                   const float* __restrict__ bias,
                   unsigned short* __restrict__ outBf,
                   float* __restrict__ outF,
                   const float* __restrict__ hres,
                   const float* __restrict__ skipPtr,
                   int skipIdx, int M, int mode) {
  __shared__ __align__(16) unsigned short ldsW[64 * 256];   // 32 KB slice
  const int colBlock = blockIdx.y;
  {
    const u32x4* g = (const u32x4*)(WT + (size_t)colBlock * 64 * 256);
    u32x4* l = (u32x4*)ldsW;
    #pragma unroll
    for (int i = 0; i < 8; ++i)
      l[threadIdx.x + i * 256] = g[threadIdx.x + i * 256];
  }
  __syncthreads();

  const int wave = threadIdx.x >> 5;
  const int lane = threadIdx.x & 31;
  const int laneHi = lane >> 4;     // 0/1 (lane halves per WMMA 16-bit layout)
  const int lanelo = lane & 15;
  const int rowBase = blockIdx.x * 128 + wave * 16;
  int aRow = rowBase + lanelo;
  if (aRow >= M) aRow = M - 1;      // clamp for load; store is guarded
  const unsigned short* aPtr = A + (size_t)aRow * 256;

  v8f acc[4] = {};
  #pragma unroll
  for (int kb = 0; kb < 8; ++kb) {
    const int k0 = kb * 32;
    Frag16 fa;
    // A 16x32 bf16 fragment: lanes 0-15 K{0..7,16..23}, lanes 16-31 K{8..15,24..31}
    fa.q[0] = *(const u32x4*)(aPtr + k0 + laneHi * 8);
    fa.q[1] = *(const u32x4*)(aPtr + k0 + 16 + laneHi * 8);
    #pragma unroll
    for (int n = 0; n < 4; ++n) {
      Frag16 fb;  // B 32x16: column (n*16+lanelo), same K split (W stored col-major)
      const unsigned short* w = ldsW + (size_t)(n * 16 + lanelo) * 256 + k0 + laneHi * 8;
      fb.q[0] = *(const u32x4*)(w);
      fb.q[1] = *(const u32x4*)(w + 16);
      acc[n] = __builtin_amdgcn_wmma_f32_16x16x32_bf16(
          false, fa.v, false, fb.v, (short)0, acc[n], false, false);
    }
  }

  float alpha = 0.0f;
  if (mode == 1) {
    float s = skipPtr[skipIdx];
    alpha = 1.0f / (1.0f + __expf(-s));
  }
  #pragma unroll
  for (int n = 0; n < 4; ++n) {
    const int col = colBlock * 64 + n * 16 + lanelo;
    const float bcol = bias[col];
    #pragma unroll
    for (int i = 0; i < 8; ++i) {
      const int row = rowBase + laneHi * 8 + i;   // C/D layout: lanes>=16 hold M+8
      if (row < M) {
        float val = acc[n][i] + bcol;
        size_t idx = (size_t)row * 256 + col;
        if (mode == 0) outBf[idx] = f2bf(val);
        else outF[idx] = alpha * val + (1.0f - alpha) * hres[idx];
      }
    }
  }
}

// ---------------------------------------------------------------------------
// Fill helpers
// ---------------------------------------------------------------------------
__global__ void hgt_fill_u32(unsigned* p, unsigned v, int n) {
  int i = blockIdx.x * blockDim.x + threadIdx.x;
  if (i < n) p[i] = v;
}
__global__ void hgt_fill_f32(float* p, float v, int n) {
  int i = blockIdx.x * blockDim.x + threadIdx.x;
  if (i < n) p[i] = v;
}

// ---------------------------------------------------------------------------
// Edge pass A: one wave32 per (edge, head).
//   score = dot(q[dst,h], k[src,h]) * pri[h] / 8 ; segment max via atomicMax
// ---------------------------------------------------------------------------
__global__ __launch_bounds__(256)
void hgt_edge_score(const unsigned short* __restrict__ Kbf,
                    const unsigned short* __restrict__ Qbf,
                    const int* __restrict__ src, const int* __restrict__ dst,
                    const float* __restrict__ relPri,
                    float* __restrict__ score,
                    unsigned* __restrict__ maxKey, int E) {
  int gw = (blockIdx.x * 256 + threadIdx.x) >> 5;
  int lane = threadIdx.x & 31;
  if (gw >= E * H) return;
  int e = gw >> 2, h = gw & 3;
  int s = src[e], d = dst[e];
  size_t qb = (size_t)d * 256 + h * 64;
  size_t kb = (size_t)s * 256 + h * 64;
  float p = bf2f(Qbf[qb + lane]) * bf2f(Kbf[kb + lane])
          + bf2f(Qbf[qb + lane + 32]) * bf2f(Kbf[kb + lane + 32]);
  #pragma unroll
  for (int off = 16; off > 0; off >>= 1) p += __shfl_xor(p, off, 32);
  if (lane == 0) {
    float sc = p * relPri[h] * 0.125f;   // 1/sqrt(DK)
    score[(size_t)e * H + h] = sc;
    atomicMax(&maxKey[(size_t)d * H + h], fkey(sc));
  }
}

// ---------------------------------------------------------------------------
// Edge pass B: e = exp(score - max[dst]); denom += e; agg[dst] += e * v[src]
// (normalization deferred to node-level finalize)
// ---------------------------------------------------------------------------
__global__ __launch_bounds__(256)
void hgt_edge_agg(const unsigned short* __restrict__ Vbf,
                  const int* __restrict__ src, const int* __restrict__ dst,
                  const float* __restrict__ score,
                  const unsigned* __restrict__ maxKey,
                  float* __restrict__ denom,
                  float* __restrict__ agg, int E) {
  int gw = (blockIdx.x * 256 + threadIdx.x) >> 5;
  int lane = threadIdx.x & 31;
  if (gw >= E * H) return;
  int e = gw >> 2, h = gw & 3;
  int s = src[e], d = dst[e];
  float m = funkey(maxKey[(size_t)d * H + h]);
  float ex = __expf(score[(size_t)e * H + h] - m);
  if (lane == 0) atomicAdd(&denom[(size_t)d * H + h], ex);
  size_t vb = (size_t)s * 256 + h * 64;
  size_t ab = (size_t)d * 256 + h * 64;
  atomicAdd(&agg[ab + lane],      ex * bf2f(Vbf[vb + lane]));
  atomicAdd(&agg[ab + lane + 32], ex * bf2f(Vbf[vb + lane + 32]));
}

// ---------------------------------------------------------------------------
// Node finalize: t = aggX/denX (and mean with aggY/denY if dual), as bf16
// ---------------------------------------------------------------------------
__global__ void hgt_finalize(const float* __restrict__ aggX, const float* __restrict__ denX,
                             const float* __restrict__ aggY, const float* __restrict__ denY,
                             unsigned short* __restrict__ tOut, int total, int dual) {
  int i = blockIdx.x * blockDim.x + threadIdx.x;
  if (i >= total) return;
  int node = i >> 8;
  int h = (i >> 6) & 3;
  float d0 = denX[(size_t)node * H + h];
  float v = (d0 > 0.0f) ? aggX[i] / d0 : 0.0f;
  if (dual) {
    float d1 = denY[(size_t)node * H + h];
    float v1 = (d1 > 0.0f) ? aggY[i] / d1 : 0.0f;
    v = 0.5f * (v + v1);
  }
  tOut[i] = f2bf(v);
}

// ---------------------------------------------------------------------------
// Host launcher
// ---------------------------------------------------------------------------
extern "C" void kernel_launch(void* const* d_in, const int* in_sizes, int n_in,
                              void* d_out, int out_size, void* d_ws, size_t ws_size,
                              hipStream_t stream) {
  (void)n_in; (void)out_size; (void)ws_size;
  const float* hA = (const float*)d_in[0];
  const float* hB = (const float*)d_in[1];
  const int* srcs[3] = {(const int*)d_in[2], (const int*)d_in[4], (const int*)d_in[6]};
  const int* dsts[3] = {(const int*)d_in[3], (const int*)d_in[5], (const int*)d_in[7]};
  const float* Wk = (const float*)d_in[8];
  const float* bk = (const float*)d_in[9];
  const float* Wq = (const float*)d_in[10];
  const float* bq = (const float*)d_in[11];
  const float* Wv = (const float*)d_in[12];
  const float* bv = (const float*)d_in[13];
  const float* Wa = (const float*)d_in[14];
  const float* ba = (const float*)d_in[15];
  const float* rel_att = (const float*)d_in[16];
  const float* rel_msg = (const float*)d_in[17];
  const float* rel_pri = (const float*)d_in[18];
  const float* skip = (const float*)d_in[19];

  const int N = in_sizes[0] / 256;
  const int E = in_sizes[2];

  // ---- workspace layout ----
  char* p = (char*)d_ws;
  auto carve = [&](size_t bytes) -> void* {
    char* r = p; p += (bytes + 255) & ~(size_t)255; return (void*)r;
  };
  const size_t nodeBf = (size_t)N * 256 * 2;
  unsigned short* hAbf = (unsigned short*)carve(nodeBf);
  unsigned short* hBbf = (unsigned short*)carve(nodeBf);
  unsigned short* WkF  = (unsigned short*)carve(3 * 65536 * 2);
  unsigned short* WvF  = (unsigned short*)carve(3 * 65536 * 2);
  unsigned short* WqT  = (unsigned short*)carve(2 * 65536 * 2);
  unsigned short* WaT  = (unsigned short*)carve(2 * 65536 * 2);
  float* bkF = (float*)carve(3 * 256 * 4);
  float* bvF = (float*)carve(3 * 256 * 4);
  unsigned short* Kb[3], *Vb[3], *Qb[2];
  for (int r = 0; r < 3; ++r) Kb[r] = (unsigned short*)carve(nodeBf);
  for (int r = 0; r < 3; ++r) Vb[r] = (unsigned short*)carve(nodeBf);
  for (int t = 0; t < 2; ++t) Qb[t] = (unsigned short*)carve(nodeBf);
  float* scorebuf = (float*)carve((size_t)E * H * 4);
  unsigned* maxbuf = (unsigned*)carve((size_t)N * H * 4);
  float* denomAll = (float*)carve((size_t)3 * N * H * 4);
  float* aggAll   = (float*)carve((size_t)3 * N * 256 * 4);
  unsigned short* tAbf = (unsigned short*)carve(nodeBf);
  unsigned short* tBbf = (unsigned short*)carve(nodeBf);

  const int nElem = N * 256;

  // ---- 1. convert node features to bf16 ----
  hgt_f2bf_kernel<<<(nElem + 255) / 256, 256, 0, stream>>>(hA, hAbf, nElem);
  hgt_f2bf_kernel<<<(nElem + 255) / 256, 256, 0, stream>>>(hB, hBbf, nElem);

  // ---- 2. fuse weights (projection @ relation transform), transposed bf16 ----
  hgt_fuse_weight_kernel<<<(3 * 65536 + 255) / 256, 256, 0, stream>>>(Wk, rel_att, bk, WkF, bkF);
  hgt_fuse_weight_kernel<<<(3 * 65536 + 255) / 256, 256, 0, stream>>>(Wv, rel_msg, bv, WvF, bvF);
  hgt_transpose_w_kernel<<<(2 * 65536) / 256, 256, 0, stream>>>(Wq, WqT);
  hgt_transpose_w_kernel<<<(2 * 65536) / 256, 256, 0, stream>>>(Wa, WaT);

  // ---- 3. WMMA GEMMs: per-relation K and V, per-type Q ----
  dim3 gblk(256);
  dim3 ggrid((N + 127) / 128, 4);
  auto gemm = [&](const unsigned short* A, const unsigned short* WT, const float* bias,
                  unsigned short* oBf, float* oF, const float* hres, int si, int mode) {
    hgt_gemm_bf16<<<ggrid, gblk, 0, stream>>>(A, WT, bias, oBf, oF, hres, skip, si, N, mode);
  };
  gemm(hAbf, WkF + 0 * 65536, bkF + 0,   Kb[0], nullptr, nullptr, 0, 0);
  gemm(hBbf, WkF + 1 * 65536, bkF + 256, Kb[1], nullptr, nullptr, 0, 0);
  gemm(hAbf, WkF + 2 * 65536, bkF + 512, Kb[2], nullptr, nullptr, 0, 0);
  gemm(hAbf, WvF + 0 * 65536, bvF + 0,   Vb[0], nullptr, nullptr, 0, 0);
  gemm(hBbf, WvF + 1 * 65536, bvF + 256, Vb[1], nullptr, nullptr, 0, 0);
  gemm(hAbf, WvF + 2 * 65536, bvF + 512, Vb[2], nullptr, nullptr, 0, 0);
  gemm(hAbf, WqT + 0 * 65536, bq + 0,    Qb[0], nullptr, nullptr, 0, 0);
  gemm(hBbf, WqT + 1 * 65536, bq + 256,  Qb[1], nullptr, nullptr, 0, 0);

  // ---- 4. zero denominators + aggregation buffers ----
  hgt_fill_f32<<<((3 * N * H) + 255) / 256, 256, 0, stream>>>(denomAll, 0.0f, 3 * N * H);
  hgt_fill_f32<<<((3 * nElem) + 255) / 256, 256, 0, stream>>>(aggAll, 0.0f, 3 * nElem);

  // ---- 5. per-relation edge softmax-aggregate (dst types: B, A, B) ----
  const unsigned short* Qr[3] = {Qb[1], Qb[0], Qb[1]};
  const int eblocks = (E * H + 7) / 8;                 // 8 wave32 per block
  for (int r = 0; r < 3; ++r) {
    hgt_fill_u32<<<((N * H) + 255) / 256, 256, 0, stream>>>(maxbuf, FKEY_NEG_INF, N * H);
    hgt_edge_score<<<eblocks, 256, 0, stream>>>(Kb[r], Qr[r], srcs[r], dsts[r],
        rel_pri + r * H, scorebuf, maxbuf, E);
    hgt_edge_agg<<<eblocks, 256, 0, stream>>>(Vb[r], srcs[r], dsts[r], scorebuf,
        maxbuf, denomAll + (size_t)r * N * H, aggAll + (size_t)r * nElem, E);
  }

  // ---- 6. normalize + cross-relation mean -> bf16 ----
  float* agg0 = aggAll;
  float* agg1 = aggAll + (size_t)1 * nElem;
  float* agg2 = aggAll + (size_t)2 * nElem;
  float* den0 = denomAll;
  float* den1 = denomAll + (size_t)1 * N * H;
  float* den2 = denomAll + (size_t)2 * N * H;
  hgt_finalize<<<(nElem + 255) / 256, 256, 0, stream>>>(agg1, den1, agg1, den1, tAbf, nElem, 0);
  hgt_finalize<<<(nElem + 255) / 256, 256, 0, stream>>>(agg0, den0, agg2, den2, tBbf, nElem, 1);

  // ---- 7. output transform + sigmoid-skip blend (WMMA, f32 out) ----
  float* outF = (float*)d_out;
  gemm(tAbf, WaT + 0 * 65536, ba + 0,   nullptr, outF,                    hA, 0, 1);
  gemm(tBbf, WaT + 1 * 65536, ba + 256, nullptr, outF + (size_t)N * 256,  hB, 1, 1);
}
